// Decoder_59261958750536
// MI455X (gfx1250) — matrix operand order repaired and checked by reading, not compiled, not run on hardware
//
#include <hip/hip_runtime.h>
#include <hip/hip_bf16.h>
#include <cstddef>

typedef __attribute__((ext_vector_type(16))) _Float16 v16h;
typedef __attribute__((ext_vector_type(8)))  float    v8f;

// ---------------- constants ----------------
#define BATCH 64
#define TENC 512
#define TMELX 700
#define MELN 80
#define MELP 96          // prenet K padded to multiple of 32
#define RR 7
#define STEPS 100
#define ATTN 128

// f16 weight offsets (in _Float16 elements) inside workspace
#define W_PRE1   0         // 256 x 96 (zero-padded from 256 x 80)
#define W_PRE2   24576     // 128 x 256
#define W_ATT_IH 57344     // 768 x 640
#define W_ATT_HH 548864    // 768 x 256
#define W_Q      745472    // 128 x 256
#define W_INP    778240    // 128 x 512
#define W_PD     843776    // 256 x 768
#define W_D1IH   1040384   // 768 x 256
#define W_D1HH   1236992
#define W_D2IH   1433600
#define W_D2HH   1630208
#define W_MEL    1826816   // 560 x 256
// f16 activation shadows (in _Float16 elements)
#define H_MIN    1970176   // 64*96 (padded)
#define H_P1     1976320   // 64*256
#define H_P2     1992704   // 64*128
#define H_X      2000896   // 64*640
#define H_HATT   2041856   // 64*256
#define H_H1     2058240   // 64*256
#define H_H2     2074624   // 64*256
#define H_DBUF   2091008   // 64*768
#define H_DEC    2140160   // 64*256
#define H_END    2156544
// byte offsets of fp32 regions
#define OFF_PROCIN 0x480000u     // 32768*128 f32 = 16 MB
#define OFF_LOCC   0x1480000u    // 64*32*512 f32 = 4 MB
#define OFF_STATE  0x1880000u
// state float offsets
#define S_HATT 0
#define S_H1   16384
#define S_H2   32768
#define S_CTX  49152
#define S_AW   81920
#define S_AWC  114688
#define S_ZEND 147456
#define S_ALGN 147456
#define S_EBUF 180224
#define S_PQ   212992
#define S_GI   221184
#define S_GH   270336
#define S_DEC  319488
#define S_OUTB 335872

#define SYNC() do { __threadfence(); __syncthreads(); } while (0)

// ------------- WMMA tile loaders (ISA 7.12.2 layouts, wave32) -------------
// All K are multiples of 32 by construction -> no bounds guards in hot loops.
// A 16x32 f16 from fp32 source (one-shot proc_in kernel only)
__device__ inline v16h load_a16(const float* __restrict__ X, int ldx, int row,
                                int k0, int kg) {
  v16h a;
  const float* xr = X + (size_t)row * ldx + k0 + kg * 8;
#pragma unroll
  for (int e = 0; e < 8; ++e) {
    a[e]     = (_Float16)xr[e];
    a[e + 8] = (_Float16)xr[16 + e];
  }
  return a;
}
// A 16x32 directly from f16 shadow: two contiguous 16B (b128) loads per lane
__device__ inline v16h load_a16h(const _Float16* __restrict__ X, int ldx, int row,
                                 int k0, int kg) {
  v16h a;
  const _Float16* xr = X + (size_t)row * ldx + k0 + kg * 8;
#pragma unroll
  for (int e = 0; e < 8; ++e) {
    a[e]     = xr[e];
    a[e + 8] = xr[16 + e];
  }
  return a;
}
// B 32x16 f16 with B[k][n] = W[n][k]: lane%16 = col n, kg picks K groups {8kg..8kg+7, 16+8kg..16+8kg+7}
__device__ inline v16h load_b16(const _Float16* __restrict__ W, int ldw, int col,
                                int k0, int kg) {
  v16h b;
  const _Float16* wr = W + (size_t)col * ldw + k0 + kg * 8;
#pragma unroll
  for (int e = 0; e < 8; ++e) {
    b[e]     = wr[e];
    b[e + 8] = wr[16 + e];
  }
  return b;
}

// Workgroup-cooperative GEMM: Y[64][N] = Xh[64][K] @ W[N][K]^T (+bias)(relu?)
// Writes fp32 Y and/or f16 shadow Yh (either may be null). K % 32 == 0.
__device__ inline void wg_gemm(const _Float16* __restrict__ X, int ldx,
                               const _Float16* __restrict__ W, int K, int N,
                               const float* __restrict__ bias,
                               float* __restrict__ Y, _Float16* __restrict__ Yh,
                               int act, int tid, int nth) {
  const int lane = tid & 31, wave = tid >> 5, nwaves = nth >> 5;
  const int row = lane & 15, kg = lane >> 4;
  const int mt = BATCH / 16, nt = N >> 4, tiles = mt * nt;
  for (int t = wave; t < tiles; t += nwaves) {
    const int m0 = (t % mt) << 4, n0 = (t / mt) << 4;
    v8f acc = {};
    for (int k0 = 0; k0 < K; k0 += 32) {
      if (k0 + 32 < K)  // overlap next weight slice fetch with WMMA issue
        __builtin_prefetch((const void*)(W + (size_t)(n0 + row) * K + k0 + 32), 0, 1);
      v16h a = load_a16h(X, ldx, m0 + row, k0, kg);
      v16h b = load_b16(W, K, n0 + row, k0, kg);
      acc = __builtin_amdgcn_wmma_f32_16x16x32_f16(false, a, false, b,
                                                   (short)0, acc, false, false);
    }
#pragma unroll
    for (int r = 0; r < 8; ++r) {
      int m = m0 + r + (kg << 3), n = n0 + row;
      float v = acc[r];
      if (bias) v += bias[n];
      if (act) v = fmaxf(v, 0.0f);
      if (Y)  Y[(size_t)m * N + n] = v;
      if (Yh) Yh[(size_t)m * N + n] = (_Float16)v;
    }
  }
}

// GRU elementwise: h' = (1-z)*tanh(inn + r*hn) + z*h ; writes f16 shadow;
// optionally dec += h' (residual) with its own f16 shadow.
__device__ inline void gru_update(float* __restrict__ h, _Float16* __restrict__ hh,
                                  const float* __restrict__ gi,
                                  const float* __restrict__ gh,
                                  float* __restrict__ dec_add,
                                  _Float16* __restrict__ dech,
                                  int tid, int nth) {
  for (int i = tid; i < BATCH * 256; i += nth) {
    int b = i >> 8, u = i & 255;
    const float* gib = gi + b * 768;
    const float* ghb = gh + b * 768;
    float r = 1.0f / (1.0f + __expf(-(gib[u] + ghb[u])));
    float z = 1.0f / (1.0f + __expf(-(gib[256 + u] + ghb[256 + u])));
    float hn = tanhf(gib[512 + u] + r * ghb[512 + u]);
    float hv = (1.0f - z) * hn + z * h[i];
    h[i] = hv;
    hh[i] = (_Float16)hv;
    if (dec_add) {
      float dv = dec_add[i] + hv;
      dec_add[i] = dv;
      dech[i] = (_Float16)dv;
    }
  }
}

// ---------------- kernels ----------------
__global__ void k_cvt_f16(const float* __restrict__ src, _Float16* __restrict__ dst, int n) {
  int i = blockIdx.x * blockDim.x + threadIdx.x;
  if (i < n) dst[i] = (_Float16)src[i];
}
// Convert with row zero-padding: src rows x Ksrc -> dst rows x Kdst (Kdst >= Ksrc)
__global__ void k_cvt_pad_f16(const float* __restrict__ src, _Float16* __restrict__ dst,
                              int rows, int Ksrc, int Kdst) {
  int i = blockIdx.x * blockDim.x + threadIdx.x;
  if (i < rows * Kdst) {
    int r = i / Kdst, k = i % Kdst;
    dst[i] = (k < Ksrc) ? (_Float16)src[(size_t)r * Ksrc + k] : (_Float16)0.0f;
  }
}

// proc_in[b*512+t][a] = sum_d inputs[b][t][d] * inp_w[a][d]   (32768 x 512 x 128, WMMA)
__global__ void k_proc_in(const float* __restrict__ inputs,
                          const _Float16* __restrict__ wh,
                          float* __restrict__ Y) {
  const int tid = threadIdx.x, lane = tid & 31;
  const int gwave = blockIdx.x * (blockDim.x >> 5) + (tid >> 5);
  const int nwaves = gridDim.x * (blockDim.x >> 5);
  const int row = lane & 15, kg = lane >> 4;
  const int MT = (BATCH * TENC) / 16;  // 2048
  const int NT = ATTN / 16;            // 8
  for (int t = gwave; t < MT * NT; t += nwaves) {
    int m0 = (t % MT) << 4, n0 = (t / MT) << 4;
    v8f acc = {};
    for (int k0 = 0; k0 < 512; k0 += 32) {
      v16h a = load_a16(inputs, 512, m0 + row, k0, kg);
      v16h b = load_b16(wh, 512, n0 + row, k0, kg);
      acc = __builtin_amdgcn_wmma_f32_16x16x32_f16(false, a, false, b,
                                                   (short)0, acc, false, false);
    }
#pragma unroll
    for (int r = 0; r < 8; ++r)
      Y[(size_t)(m0 + r + (kg << 3)) * ATTN + n0 + row] = acc[r];
  }
}

// Persistent serial-scan decoder: one workgroup, 32 wave32 waves, 100 steps.
__global__ void __launch_bounds__(1024) k_decoder(
    const float* __restrict__ inputs, const float* __restrict__ memory,
    const unsigned char* __restrict__ mask,
    const float* __restrict__ procin, float* __restrict__ locc,
    float* __restrict__ S, _Float16* __restrict__ Wh,
    const float* __restrict__ pre_b1, const float* __restrict__ pre_b2,
    const float* __restrict__ att_bih, const float* __restrict__ att_bhh,
    const float* __restrict__ vw, const float* __restrict__ vb,
    const float* __restrict__ lconv, const float* __restrict__ lw,
    const float* __restrict__ pd_b,
    const float* __restrict__ d1_bih, const float* __restrict__ d1_bhh,
    const float* __restrict__ d2_bih, const float* __restrict__ d2_bhh,
    const float* __restrict__ mel_b,
    const float* __restrict__ stop_w, const float* __restrict__ stop_b,
    float* __restrict__ out) {
  const int tid = threadIdx.x, nth = blockDim.x;

  float* hatt = S + S_HATT; float* h1 = S + S_H1;   float* h2 = S + S_H2;
  float* ctx  = S + S_CTX;  float* aw = S + S_AW;   float* awc = S + S_AWC;
  float* algn = S + S_ALGN; float* ebuf = S + S_EBUF; float* pq = S + S_PQ;
  float* gi   = S + S_GI;   float* gh = S + S_GH;
  float* dec  = S + S_DEC;  float* outb = S + S_OUTB;

  _Float16* minh  = Wh + H_MIN;  _Float16* p1h = Wh + H_P1;  _Float16* p2h = Wh + H_P2;
  _Float16* xh    = Wh + H_X;    _Float16* hatth = Wh + H_HATT;
  _Float16* h1h   = Wh + H_H1;   _Float16* h2h = Wh + H_H2;
  _Float16* dbufh = Wh + H_DBUF; _Float16* dech = Wh + H_DEC;

  float* outputs = out;                                   // 64*80*700
  float* atts    = out + (size_t)BATCH * MELN * TMELX;    // 64*100*512
  float* stops   = atts + (size_t)BATCH * STEPS * TENC;   // 64*100

  for (int i = tid; i < S_ZEND; i += nth) S[i] = 0.0f;    // zero fp32 carries
  for (int i = tid; i < 3 * BATCH * 256; i += nth)        // zero h shadows
    hatth[i] = (_Float16)0.0f;                            // hatth,h1h,h2h contiguous
  SYNC();

  for (int s = 0; s < STEPS; ++s) {
    // ---- m_in = last mel frame of previous group (f16, K padded to 96) ----
    for (int i = tid; i < BATCH * MELP; i += nth) {
      int b = i / MELP, k = i % MELP;
      minh[i] = (k < MELN && s > 0)
                    ? (_Float16)memory[(size_t)b * TMELX * MELN + (s * RR - 1) * MELN + k]
                    : (_Float16)0.0f;
    }
    SYNC();
    // ---- prenet ----
    wg_gemm(minh, MELP, Wh + W_PRE1, MELP, 256, pre_b1, nullptr, p1h, 1, tid, nth); SYNC();
    wg_gemm(p1h, 256, Wh + W_PRE2, 256, 128, pre_b2, nullptr, p2h, 1, tid, nth);    SYNC();
    // ---- x = cat(p2, ctx) (f16) ----
    for (int i = tid; i < BATCH * 640; i += nth) {
      int b = i / 640, k = i % 640;
      xh[i] = (k < 128) ? p2h[b * 128 + k] : (_Float16)ctx[b * 512 + (k - 128)];
    }
    SYNC();
    // ---- attention GRU (gh uses previous-step hatth) ----
    wg_gemm(xh, 640, Wh + W_ATT_IH, 640, 768, att_bih, gi, nullptr, 0, tid, nth);
    wg_gemm(hatth, 256, Wh + W_ATT_HH, 256, 768, att_bhh, gh, nullptr, 0, tid, nth);
    SYNC();
    gru_update(hatt, hatth, gi, gh, nullptr, nullptr, tid, nth);
    SYNC();
    // ---- query projection + location conv (independent phases) ----
    wg_gemm(hatth, 256, Wh + W_Q, 256, 128, nullptr, pq, nullptr, 0, tid, nth);
    for (int i = tid; i < BATCH * 32 * TENC; i += nth) {
      int b = i / (32 * TENC), c = (i >> 9) & 31, t0 = i & 511;
      const float* w0 = lconv + c * 62;
      float acc = 0.0f;
      for (int k = 0; k < 31; ++k) {
        int tt = t0 + k - 15;
        if (tt >= 0 && tt < TENC)
          acc += w0[k] * aw[b * TENC + tt] + w0[31 + k] * awc[b * TENC + tt];
      }
      locc[i] = acc;
    }
    SYNC();
    // ---- energies: e = v . tanh(pq + proc_in + loc_w@conv) ----
    for (int i = tid; i < BATCH * TENC; i += nth) {
      int b = i >> 9, t0 = i & 511;
      float lc[32];
#pragma unroll
      for (int c = 0; c < 32; ++c) lc[c] = locc[(size_t)(b * 32 + c) * TENC + t0];
      const float* pr = procin + (size_t)(b * TENC + t0) * ATTN;
      const float* pqb = pq + b * ATTN;
      float e = vb[0];
#pragma unroll 1
      for (int a = 0; a < ATTN; ++a) {
        const float* lwa = lw + a * 32;
        float pl = 0.0f;
#pragma unroll
        for (int c = 0; c < 32; ++c) pl += lc[c] * lwa[c];
        e += vw[a] * tanhf(pqb[a] + pr[a] + pl);
      }
      ebuf[i] = mask[i] ? e : -1.0e9f;
    }
    SYNC();
    // ---- softmax over T_ENC (16 lanes per batch row) + aw/awc/atts ----
    {
      int b = tid >> 4, sub = tid & 15;
      float m = -1.0e30f;
      for (int t0 = sub; t0 < TENC; t0 += 16) m = fmaxf(m, ebuf[b * TENC + t0]);
      for (int o = 8; o > 0; o >>= 1) m = fmaxf(m, __shfl_xor(m, o, 16));
      float sum = 0.0f;
      for (int t0 = sub; t0 < TENC; t0 += 16) sum += __expf(ebuf[b * TENC + t0] - m);
      for (int o = 8; o > 0; o >>= 1) sum += __shfl_xor(sum, o, 16);
      float inv = 1.0f / sum;
      for (int t0 = sub; t0 < TENC; t0 += 16) {
        float al = __expf(ebuf[b * TENC + t0] - m) * inv;
        algn[b * TENC + t0] = al;
        aw[b * TENC + t0] = al;
        awc[b * TENC + t0] += al;
        atts[((size_t)b * STEPS + s) * TENC + t0] = al;
      }
    }
    SYNC();
    // ---- context: ctx[b][d] = sum_t align[b][t] * inputs[b][t][d] ----
    for (int i = tid; i < BATCH * 512; i += nth) {
      int b = i >> 9, d = i & 511;
      const float* al = algn + b * TENC;
      const float* inb = inputs + (size_t)b * TENC * 512 + d;
      float acc = 0.0f;
      for (int t0 = 0; t0 < TENC; ++t0) acc += al[t0] * inb[(size_t)t0 * 512];
      ctx[i] = acc;
    }
    SYNC();
    // ---- dbuf = cat(h_att, ctx) (f16) ----
    for (int i = tid; i < BATCH * 768; i += nth) {
      int b = i / 768, k = i % 768;
      dbufh[i] = (k < 256) ? hatth[b * 256 + k] : (_Float16)ctx[b * 512 + (k - 256)];
    }
    SYNC();
    wg_gemm(dbufh, 768, Wh + W_PD, 768, 256, pd_b, dec, dech, 0, tid, nth);
    SYNC();
    // ---- decoder GRU 1 (residual) ----
    wg_gemm(dech, 256, Wh + W_D1IH, 256, 768, d1_bih, gi, nullptr, 0, tid, nth);
    wg_gemm(h1h, 256, Wh + W_D1HH, 256, 768, d1_bhh, gh, nullptr, 0, tid, nth);
    SYNC();
    gru_update(h1, h1h, gi, gh, dec, dech, tid, nth);
    SYNC();
    // ---- decoder GRU 2 (residual) ----
    wg_gemm(dech, 256, Wh + W_D2IH, 256, 768, d2_bih, gi, nullptr, 0, tid, nth);
    wg_gemm(h2h, 256, Wh + W_D2HH, 256, 768, d2_bhh, gh, nullptr, 0, tid, nth);
    SYNC();
    gru_update(h2, h2h, gi, gh, dec, dech, tid, nth);
    SYNC();
    // ---- mel projection ----
    wg_gemm(dech, 256, Wh + W_MEL, 256, MELN * RR, mel_b, outb, nullptr, 0, tid, nth);
    SYNC();
    // ---- scatter outputs, stop token ----
    for (int i = tid; i < BATCH * MELN * RR; i += nth) {
      int b = i / (MELN * RR), j = i % (MELN * RR);
      int r = j / MELN, m = j % MELN;
      outputs[(size_t)b * MELN * TMELX + (size_t)m * TMELX + s * RR + r] = outb[i];
    }
    if (tid < BATCH) {
      int b = tid;
      float acc = stop_b[0];
      for (int k = 0; k < 256; ++k) acc += dec[b * 256 + k] * stop_w[k];
      for (int j = 0; j < MELN * RR; ++j) acc += outb[b * (MELN * RR) + j] * stop_w[256 + j];
      stops[b * STEPS + s] = acc;
    }
    SYNC();
  }
}

// ---------------- launch ----------------
extern "C" void kernel_launch(void* const* d_in, const int* in_sizes, int n_in,
                              void* d_out, int out_size, void* d_ws, size_t ws_size,
                              hipStream_t stream) {
  (void)in_sizes; (void)n_in; (void)out_size; (void)ws_size;
  char* base = (char*)d_ws;
  _Float16* Wh = (_Float16*)base;
  float* procin = (float*)(base + OFF_PROCIN);
  float* locc   = (float*)(base + OFF_LOCC);
  float* S      = (float*)(base + OFF_STATE);

  auto cvt = [&](int idx, int off, int n) {
    k_cvt_f16<<<(n + 255) / 256, 256, 0, stream>>>((const float*)d_in[idx], Wh + off, n);
  };
  // prenet_w1 zero-padded 256x80 -> 256x96
  k_cvt_pad_f16<<<(256 * MELP + 255) / 256, 256, 0, stream>>>(
      (const float*)d_in[3], Wh + W_PRE1, 256, MELN, MELP);
  cvt(5,  W_PRE2,   128 * 256);
  cvt(7,  W_ATT_IH, 768 * 640);
  cvt(8,  W_ATT_HH, 768 * 256);
  cvt(11, W_Q,      128 * 256);
  cvt(12, W_INP,    128 * 512);
  cvt(17, W_PD,     256 * 768);
  cvt(19, W_D1IH,   768 * 256);
  cvt(20, W_D1HH,   768 * 256);
  cvt(23, W_D2IH,   768 * 256);
  cvt(24, W_D2HH,   768 * 256);
  cvt(27, W_MEL,    560 * 256);

  k_proc_in<<<512, 256, 0, stream>>>((const float*)d_in[0], Wh + W_INP, procin);

  k_decoder<<<1, 1024, 0, stream>>>(
      (const float*)d_in[0], (const float*)d_in[1], (const unsigned char*)d_in[2],
      procin, locc, S, Wh,
      (const float*)d_in[4],  (const float*)d_in[6],
      (const float*)d_in[9],  (const float*)d_in[10],
      (const float*)d_in[13], (const float*)d_in[14],
      (const float*)d_in[15], (const float*)d_in[16],
      (const float*)d_in[18],
      (const float*)d_in[21], (const float*)d_in[22],
      (const float*)d_in[25], (const float*)d_in[26],
      (const float*)d_in[28], (const float*)d_in[29], (const float*)d_in[30],
      (float*)d_out);
}
